// AGSPN_60421599920502
// MI455X (gfx1250) — compile-verified, matrix operand order checked by reading, not run on hardware
//
#include <hip/hip_runtime.h>
#include <math.h>

// ---------------- problem constants ----------------
constexpr int Bc  = 4;
constexpr int Hc  = 240;
constexpr int Wc  = 1216;
constexpr int HWc = Hc * Wc;
constexpr int PROP = 6;
constexpr int TILE = 128;   // pixels per block in affinity kernel (8 waves x 16 px)
constexpr int T3   = 64;    // pixels per block in post kernel
constexpr int GSM_N = 8 * 3 * 130;   // guidance tile floats
constexpr int ZSLOT = GSM_N;         // zero slot for padded-K reads

// ---------------- vector types ----------------
typedef __attribute__((ext_vector_type(16))) _Float16     v16h;
typedef __attribute__((ext_vector_type(8)))  float        v8f;
typedef __attribute__((ext_vector_type(4)))  unsigned int v4u;
typedef __attribute__((ext_vector_type(8)))  int          v8i;
typedef __attribute__((ext_vector_type(4)))  int          v4i;

union V16H { v16h v; _Float16 e[16]; };
union V8F  { v8f  v; float    e[8];  };

// K index held by lane `lane` at fragment slot i (16-bit A/B layout, 16x16x32)
__device__ __forceinline__ int kmap(int lane, int i) {
    // lanes 0-15:  K = i      (i<8)  ; i+8  (i>=8)
    // lanes 16-31: K = i+8    (i<8)  ; i+16 (i>=8)
    return i + ((lane < 16) ? (i >= 8 ? 8 : 0) : (i >= 8 ? 16 : 8));
}

__device__ __forceinline__ float sigmoidf_(float v) {
    return 1.0f / (1.0f + __expf(-v));
}

// =====================================================================
// Kernel 1: fused affinity (WMMA 25x72 GEMM per pixel) + deformable conv
// grid: (ceil(W/128), H, B), block: 256 (8 waves, 16 px each)
// =====================================================================
__global__ __launch_bounds__(256)
void aff_prop_kernel(const float* __restrict__ guid,     // guidance + 8k*HW
                     const float* __restrict__ featIn,   // (B,H,W)
                     float*       __restrict__ featOut,  // (B,H,W)
                     const float* __restrict__ ww,       // (9,8,3,3)
                     const float* __restrict__ wb,       // (9)
                     const float* __restrict__ ow,       // (16,8,3,3)
                     const float* __restrict__ ob,       // (16)
                     const float* __restrict__ w3,       // (9)
                     const float* __restrict__ b3)       // (1)
{
    __shared__ float    gsm[GSM_N + 1];     // [ch][row][col] TDM-packed + zero slot
    __shared__ _Float16 wsm[32 * 96];       // weights, A-matrix padded
    __shared__ float    resm[TILE][33];     // GEMM results per pixel

    const int tid = threadIdx.x;
    const int x0  = blockIdx.x * TILE;
    const int y   = blockIdx.y;
    const int b   = blockIdx.z;

    // ---- issue TDM first (interior tiles) so it overlaps weight staging ----
    const bool interior = (x0 > 0) && (y > 0);
    if (interior && tid < 32) {   // wave 0 only: EXEC ignored by TDM
        const float* gp = guid + (size_t)(b * 48) * HWc
                               + (size_t)(y - 1) * Wc + (x0 - 1);
        unsigned long long ga = (unsigned long long)(const void*)gp;
        unsigned ldsa = (unsigned)(unsigned long long)(void*)gsm;

        // D# group 0: count=1 | lds_addr | global_addr | type=2
        v4u g0 = { 1u, ldsa, (unsigned)ga,
                   (unsigned)((ga >> 32) & 0x1FFFFFFull) | (2u << 30) };

        unsigned tdim0 = (unsigned)(Wc - (x0 - 1));   // right-edge OOB -> 0
        unsigned tdim1 = (unsigned)(Hc - (y - 1));    // bottom-edge OOB -> 0
        unsigned hw    = (unsigned)HWc;
        // D# group 1
        int e0 = (int)(2u << 16);                                   // data_size = 4B
        int e1 = (int)((tdim0 & 0xFFFFu) << 16);                    // tensor_dim0 lo
        int e2 = (int)(((tdim0 >> 16) & 0xFFFFu) | ((tdim1 & 0xFFFFu) << 16));
        int e3 = (int)(((tdim1 >> 16) & 0xFFFFu) | (130u << 16));   // tile_dim0=130
        int e4 = (int)(3u | (8u << 16));                            // tile_dim1=3, tile_dim2=8
        int e5 = (int)Wc;                                           // dim0_stride = W
        int e6 = (int)((hw & 0xFFFFu) << 16);                       // dim1_stride lo16 (=H*W)
        int e7 = (int)(hw >> 16);                                   // dim1_stride hi
        v8i g1 = { e0, e1, e2, e3, e4, e5, e6, e7 };
        v4i g2 = { 8, 0, 0, 0 };                                    // tensor_dim2 = 8
        v4i g3 = { 0, 0, 0, 0 };
        v8i gz = { 0, 0, 0, 0, 0, 0, 0, 0 };
        __builtin_amdgcn_tensor_load_to_lds(g0, g1, g2, g3, gz, 0);
    }
    if (tid == 0) gsm[ZSLOT] = 0.0f;        // zero slot for padded K reads

    // ---- stage weights into LDS as fp16 (32 rows x 96 K, zero padded) ----
    for (int idx = tid; idx < 32 * 96; idx += 256) {
        int m = idx / 96, K = idx % 96;
        float v = 0.0f;
        if (K < 72 && m < 25) {
            int kk = K >> 3, ci = K & 7;            // K = (ky*3+kx)*8 + ci
            if (m < 9) v = ww[(m * 8 + ci) * 9 + kk];
            else       v = ow[((m - 9) * 8 + ci) * 9 + kk];
        }
        wsm[idx] = (_Float16)v;
    }

    // ---- edge tiles: cooperative load with explicit zero padding ----
    if (!interior) {
        for (int idx = tid; idx < GSM_N; idx += 256) {
            int z = idx / 390, rem = idx % 390;
            int r = rem / 130, c = rem % 130;
            int gy = y - 1 + r, gx = x0 - 1 + c;
            float v = 0.0f;
            if (gy >= 0 && gy < Hc && gx >= 0 && gx < Wc)
                v = guid[(size_t)(b * 48 + z) * HWc + (size_t)gy * Wc + gx];
            gsm[idx] = v;
        }
    }
    if (interior && tid < 32) __builtin_amdgcn_s_wait_tensorcnt(0);
    __syncthreads();

    // ---- build A fragments (weights), two 16-channel tiles, 3 K-steps ----
    const int lane = tid & 31;
    const int wave = tid >> 5;
    const int m0   = lane & 15;

    V16H afrag[3][2];
    for (int s = 0; s < 3; ++s)
        for (int t = 0; t < 2; ++t) {
            int m = m0 + 16 * t;
#pragma unroll
            for (int i = 0; i < 16; ++i)
                afrag[s][t].e[i] = wsm[m * 96 + (s * 32 + kmap(lane, i))];
        }

    // ---- K loop: build B from im2col of LDS tile, accumulate with WMMA ----
    const int local = wave * 16 + m0;        // pixel column within tile (N = m0)
    V8F acc0, acc1;
#pragma unroll
    for (int i = 0; i < 8; ++i) { acc0.e[i] = 0.0f; acc1.e[i] = 0.0f; }

    for (int s = 0; s < 3; ++s) {
        V16H bfrag;
#pragma unroll
        for (int i = 0; i < 16; ++i) {
            int K = s * 32 + kmap(lane, i);
            int kk = K >> 3, ci = K & 7;
            int ky = kk / 3, kx = kk - ky * 3;
            int idx = (ci * 3 + ky) * 130 + local + kx;
            idx = (K < 72) ? idx : ZSLOT;    // address select, no branch
            bfrag.e[i] = (_Float16)gsm[idx];
        }
        acc0.v = __builtin_amdgcn_wmma_f32_16x16x32_f16(
            false, afrag[s][0].v, false, bfrag.v, (short)0, acc0.v, false, false);
        acc1.v = __builtin_amdgcn_wmma_f32_16x16x32_f16(
            false, afrag[s][1].v, false, bfrag.v, (short)0, acc1.v, false, false);
    }

    // ---- scatter D tiles (16x16 f32 layout) to per-pixel LDS rows ----
    {
        int px = wave * 16 + m0;
        int cb = (lane < 16) ? 0 : 8;
#pragma unroll
        for (int r = 0; r < 8; ++r) {
            resm[px][cb + r]      = acc0.e[r];   // channels 0..15
            resm[px][16 + cb + r] = acc1.e[r];   // channels 16..31 (use 16..24)
        }
    }
    __syncthreads();

    // ---- epilogue: sigmoid/normalize weights, offsets, deformable gather ----
    if (tid < TILE) {
        int x = x0 + tid;
        if (x < Wc) {
            float wgt[9], sum = 0.0f;
#pragma unroll
            for (int j = 0; j < 9; ++j) {
                float v = sigmoidf_(resm[tid][j] + wb[j]);
                wgt[j] = v; sum += v;
            }
            float inv = 1.0f / (sum + 1e-8f);

            float ov[16];
#pragma unroll
            for (int t = 0; t < 16; ++t) ov[t] = resm[tid][9 + t] + ob[t];

            const float* img = featIn + (size_t)b * HWc;
            float acc = 0.0f;
#pragma unroll
            for (int kk = 0; kk < 9; ++kk) {
                float oy, ox;
                if (kk < 4)       { oy = ov[2 * kk];       ox = ov[2 * kk + 1]; }
                else if (kk == 4) { oy = 0.0f;             ox = 0.0f; }
                else              { oy = ov[2 * (kk - 1)]; ox = ov[2 * (kk - 1) + 1]; }
                int ky = kk / 3, kx = kk - ky * 3;
                float py  = (float)(y - 1 + ky) + oy;
                float pxp = (float)(x - 1 + kx) + ox;

                float yf = floorf(py), xf = floorf(pxp);
                int yi = (int)yf, xi = (int)xf;
                float fy = py - yf, fx = pxp - xf;
                float sv = 0.0f;
#pragma unroll
                for (int dy = 0; dy < 2; ++dy)
#pragma unroll
                    for (int dx = 0; dx < 2; ++dx) {
                        int yy = yi + dy, xx = xi + dx;
                        if (yy >= 0 && yy < Hc && xx >= 0 && xx < Wc) {
                            float wv = (dy ? fy : 1.0f - fy) * (dx ? fx : 1.0f - fx);
                            sv += wv * img[(size_t)yy * Wc + xx];
                        }
                    }
                acc += wgt[kk] * inv * sv * w3[kk];
            }
            featOut[(size_t)b * HWc + (size_t)y * Wc + x] = acc + b3[0];
        }
    }
}

// =====================================================================
// Kernel 2a/2b: batch-norm statistics (sum, sumsq of proj output, 6 ch)
// =====================================================================
__global__ void zero_stats(float* s) {
    if (threadIdx.x < 12) s[threadIdx.x] = 0.0f;
}

__global__ __launch_bounds__(256)
void stats_kernel(const float* __restrict__ f3, const float* __restrict__ f4,
                  const float* __restrict__ f5, const float* __restrict__ projw,
                  float* __restrict__ stats)
{
    __shared__ float acc[12];
    if (threadIdx.x < 12) acc[threadIdx.x] = 0.0f;
    __syncthreads();

    float l[12];
#pragma unroll
    for (int i = 0; i < 12; ++i) l[i] = 0.0f;

    const size_t N = (size_t)Bc * HWc;
    for (size_t p = (size_t)blockIdx.x * 256 + threadIdx.x; p < N;
         p += (size_t)gridDim.x * 256) {
        float y0 = f3[p], y1 = f4[p], y2 = f5[p];
#pragma unroll
        for (int o = 0; o < 6; ++o) {
            float sf = projw[o * 3] * y0 + projw[o * 3 + 1] * y1 + projw[o * 3 + 2] * y2;
            l[o] += sf; l[6 + o] += sf * sf;
        }
    }
#pragma unroll
    for (int i = 0; i < 12; ++i) atomicAdd(&acc[i], l[i]);
    __syncthreads();
    if (threadIdx.x < 12) atomicAdd(&stats[threadIdx.x], acc[threadIdx.x]);
}

// =====================================================================
// Kernel 3: fused post-processing (proj/bn/leaky -> grouped & dense convs
//           -> channel mean/max -> sigmoid gate -> cv -> dot with y)
// grid: (ceil(W/T3), H, B), block 256
// =====================================================================
__global__ __launch_bounds__(256)
void post_kernel(const float* __restrict__ f3, const float* __restrict__ f4,
                 const float* __restrict__ f5, const float* __restrict__ attn,
                 const float* __restrict__ projw,
                 const float* __restrict__ bng, const float* __restrict__ bnb,
                 const float* __restrict__ c0w, const float* __restrict__ c0b,
                 const float* __restrict__ csw, const float* __restrict__ csb,
                 const float* __restrict__ c1w, const float* __restrict__ c1b,
                 const float* __restrict__ c2w, const float* __restrict__ c2b,
                 const float* __restrict__ csqw, const float* __restrict__ csqb,
                 const float* __restrict__ cvw, const float* __restrict__ cvb,
                 const float* __restrict__ stats, float* __restrict__ out)
{
    __shared__ float a1sm[5][T3 + 4][22];   // a1 (post-c0) 5-row window, halo 2
    __shared__ float aggsm[3][T3 + 4][2];   // [mean,max] window, halo 1
    __shared__ float Wp[18], Bg[6], Bb2[6], Mu[6], Is[6];
    __shared__ float C0w[22], C0b[22], Csw[198], Csb[22];
    __shared__ float C1w[242], C1b[11], C2w[242], C2b[11];
    __shared__ float Qw[36], Qb[2], Vw[33], Vb[3];

    const int tid = threadIdx.x;
    const int x0  = blockIdx.x * T3;
    const int y   = blockIdx.y;
    const int b   = blockIdx.z;

    // ---- stage weights/constants ----
    for (int i = tid; i < 18;  i += 256) Wp[i]  = projw[i];
    for (int i = tid; i < 22;  i += 256) { C0w[i] = c0w[i]; C0b[i] = c0b[i];
                                           Csb[i] = csb[i]; }
    for (int i = tid; i < 198; i += 256) Csw[i] = csw[i];
    for (int i = tid; i < 242; i += 256) { C1w[i] = c1w[i]; C2w[i] = c2w[i]; }
    for (int i = tid; i < 11;  i += 256) { C1b[i] = c1b[i]; C2b[i] = c2b[i]; }
    for (int i = tid; i < 36;  i += 256) Qw[i] = csqw[i];
    for (int i = tid; i < 33;  i += 256) Vw[i] = cvw[i];
    if (tid < 2) Qb[tid] = csqb[tid];
    if (tid < 3) Vb[tid] = cvb[tid];
    if (tid < 6) {
        float Nf = (float)((size_t)Bc * HWc);
        float mu  = stats[tid] / Nf;
        float var = stats[6 + tid] / Nf - mu * mu;
        Mu[tid] = mu;
        Is[tid] = rsqrtf(var + 1e-5f);
        Bg[tid] = bng[tid]; Bb2[tid] = bnb[tid];
    }
    __syncthreads();

    // ---- a1 = c0(concat(attn, leaky(bn(proj(y))))) over 5x(T3+4) window ----
    for (int idx = tid; idx < 5 * (T3 + 4); idx += 256) {
        int r = idx / (T3 + 4), c = idx % (T3 + 4);
        int gy = y - 2 + r, gx = x0 - 2 + c;
        float a1[22];
        if (gy >= 0 && gy < Hc && gx >= 0 && gx < Wc) {
            size_t p = (size_t)b * HWc + (size_t)gy * Wc + gx;
            float y0 = f3[p], y1 = f4[p], y2 = f5[p];
#pragma unroll
            for (int o = 0; o < 6; ++o) {
                float sf = Wp[o * 3] * y0 + Wp[o * 3 + 1] * y1 + Wp[o * 3 + 2] * y2;
                sf = (sf - Mu[o]) * Is[o];
                sf = sf * Bg[o] + Bb2[o];
                sf = sf > 0.0f ? sf : 0.2f * sf;
                a1[16 + o] = sf * C0w[16 + o] + C0b[16 + o];
            }
            const float* ap = attn + (size_t)b * 16 * HWc + (size_t)gy * Wc + gx;
#pragma unroll
            for (int ch = 0; ch < 16; ++ch)
                a1[ch] = ap[(size_t)ch * HWc] * C0w[ch] + C0b[ch];
        } else {
#pragma unroll
            for (int ch = 0; ch < 22; ++ch) a1[ch] = 0.0f;
        }
#pragma unroll
        for (int ch = 0; ch < 22; ++ch) a1sm[r][c][ch] = a1[ch];
    }
    __syncthreads();

    // ---- agg = [mean_c, max_c] of concat(c1(a1), c2(cs(a1))) ----
    for (int idx = tid; idx < 3 * (T3 + 2); idx += 256) {
        int r = idx / (T3 + 2), c = idx % (T3 + 2) + 1;
        int gy = y - 1 + r, gx = x0 - 2 + c;
        float mean = 0.0f, mx = -3.4e38f;
        if (gy >= 0 && gy < Hc && gx >= 0 && gx < Wc) {
            float a2[22];
#pragma unroll
            for (int ch = 0; ch < 22; ++ch) {
                float s = Csb[ch];
#pragma unroll
                for (int dy = 0; dy < 3; ++dy)
#pragma unroll
                    for (int dx = 0; dx < 3; ++dx)
                        s += Csw[ch * 9 + dy * 3 + dx] * a1sm[r + dy][c - 1 + dx][ch];
                a2[ch] = s;
            }
#pragma unroll
            for (int j = 0; j < 11; ++j) {
                float s1 = C1b[j], s2 = C2b[j];
#pragma unroll
                for (int ch = 0; ch < 22; ++ch) {
                    s1 += C1w[j * 22 + ch] * a1sm[r + 1][c][ch];
                    s2 += C2w[j * 22 + ch] * a2[ch];
                }
                mean += s1 + s2;
                mx = fmaxf(mx, fmaxf(s1, s2));
            }
            mean *= (1.0f / 22.0f);
        } else {
            mean = 0.0f; mx = 0.0f;   // csq zero-pads agg outside the image
        }
        aggsm[r][c][0] = mean;
        aggsm[r][c][1] = mx;
    }
    if (tid < 3) {
        aggsm[tid][0][0] = 0.0f; aggsm[tid][0][1] = 0.0f;
        aggsm[tid][T3 + 3][0] = 0.0f; aggsm[tid][T3 + 3][1] = 0.0f;
    }
    __syncthreads();

    // ---- final: sigmoid gate, blend, cv, dot with y ----
    if (tid < T3) {
        int x = x0 + tid;
        if (x < Wc) {
            int c = tid + 2;
            float s0 = Qb[0], s1 = Qb[1];
#pragma unroll
            for (int i = 0; i < 2; ++i)
#pragma unroll
                for (int dy = 0; dy < 3; ++dy)
#pragma unroll
                    for (int dx = 0; dx < 3; ++dx) {
                        float v = aggsm[dy][c - 1 + dx][i];
                        s0 += Qw[(0 * 2 + i) * 9 + dy * 3 + dx] * v;
                        s1 += Qw[(1 * 2 + i) * 9 + dy * 3 + dx] * v;
                    }
            s0 = sigmoidf_(s0);
            s1 = sigmoidf_(s1);

            float a2[22];
#pragma unroll
            for (int ch = 0; ch < 22; ++ch) {
                float s = Csb[ch];
#pragma unroll
                for (int dy = 0; dy < 3; ++dy)
#pragma unroll
                    for (int dx = 0; dx < 3; ++dx)
                        s += Csw[ch * 9 + dy * 3 + dx] * a1sm[1 + dy][c - 1 + dx][ch];
                a2[ch] = s;
            }
            float av[3] = { Vb[0], Vb[1], Vb[2] };
#pragma unroll
            for (int j = 0; j < 11; ++j) {
                float s1c = C1b[j], s2c = C2b[j];
#pragma unroll
                for (int ch = 0; ch < 22; ++ch) {
                    s1c += C1w[j * 22 + ch] * a1sm[2][c][ch];
                    s2c += C2w[j * 22 + ch] * a2[ch];
                }
                float aj = s1c * s0 + s2c * s1;
#pragma unroll
                for (int o = 0; o < 3; ++o) av[o] += Vw[o * 11 + j] * aj;
            }
            size_t p = (size_t)b * HWc + (size_t)y * Wc + x;
            out[p] = f3[p] * av[0] + f4[p] * av[1] + f5[p] * av[2];
        }
    }
}

// =====================================================================
// host launcher
// =====================================================================
extern "C" void kernel_launch(void* const* d_in, const int* in_sizes, int n_in,
                              void* d_out, int out_size, void* d_ws, size_t ws_size,
                              hipStream_t stream)
{
    (void)in_sizes; (void)n_in; (void)out_size; (void)ws_size;

    const float* feat_init = (const float*)d_in[0];
    const float* guidance  = (const float*)d_in[1];
    const float* attn      = (const float*)d_in[2];
    // d_in[3] = feat_fix (unused by reference output)
    const float* aff_w_w   = (const float*)d_in[4];
    const float* aff_w_b   = (const float*)d_in[5];
    const float* aff_o_w   = (const float*)d_in[6];
    const float* aff_o_b   = (const float*)d_in[7];
    const float* w3        = (const float*)d_in[8];
    const float* b3        = (const float*)d_in[9];
    const float* proj_w    = (const float*)d_in[10];
    const float* bn_g      = (const float*)d_in[11];
    const float* bn_b      = (const float*)d_in[12];
    const float* c0_w      = (const float*)d_in[13];
    const float* c0_b      = (const float*)d_in[14];
    const float* cs_w      = (const float*)d_in[15];
    const float* cs_b      = (const float*)d_in[16];
    const float* c1_w      = (const float*)d_in[17];
    const float* c1_b      = (const float*)d_in[18];
    const float* c2_w      = (const float*)d_in[19];
    const float* c2_b      = (const float*)d_in[20];
    const float* csq_w     = (const float*)d_in[21];
    const float* csq_b     = (const float*)d_in[22];
    const float* cv_w      = (const float*)d_in[23];
    const float* cv_b      = (const float*)d_in[24];

    const size_t N = (size_t)Bc * HWc;
    float* feats = (float*)d_ws;          // 6 * N floats
    float* stats = feats + 6 * N;         // 12 floats

    dim3 blkA(256), grdA((Wc + TILE - 1) / TILE, Hc, Bc);
    const float* fin = feat_init;
    for (int k = 0; k < PROP; ++k) {
        float* fout = feats + (size_t)k * N;
        aff_prop_kernel<<<grdA, blkA, 0, stream>>>(
            guidance + (size_t)8 * k * HWc, fin, fout,
            aff_w_w + (size_t)k * 648, aff_w_b + (size_t)k * 9,
            aff_o_w + (size_t)k * 1152, aff_o_b + (size_t)k * 16, w3, b3);
        fin = fout;
    }

    zero_stats<<<1, 32, 0, stream>>>(stats);
    stats_kernel<<<2048, 256, 0, stream>>>(feats + 3 * N, feats + 4 * N,
                                           feats + 5 * N, proj_w, stats);

    dim3 grdP((Wc + T3 - 1) / T3, Hc, Bc);
    post_kernel<<<grdP, 256, 0, stream>>>(
        feats + 3 * N, feats + 4 * N, feats + 5 * N, attn,
        proj_w, bn_g, bn_b, c0_w, c0_b, cs_w, cs_b,
        c1_w, c1_b, c2_w, c2_b, csq_w, csq_b, cv_w, cv_b,
        stats, (float*)d_out);
}
